// GNN_modules_28879360098864
// MI455X (gfx1250) — matrix operand order repaired and checked by reading, not compiled, not run on hardware
//
#include <hip/hip_runtime.h>
#include <hip/hip_bf16.h>

typedef _Float16 f16;
typedef __attribute__((ext_vector_type(16))) _Float16 v16h;
typedef __attribute__((ext_vector_type(8)))  float    v8f;

#define NN      50000
#define NE      400000
#define FEATS   64
#define MDIM    16
#define EIN     133      // 2*64+4+1
#define EHID    266
#define NLAYERS 3

// tile counts (16x16x32 WMMA)
#define KT1 5    // ceil(133/32)  -> pad 160
#define NT1 17   // ceil(266/16)  -> pad 272
#define KT2 9    // ceil(266/32)  -> pad 288
#define KT3 1    // 16 -> pad 32
#define NT3 4    // 64
#define NKT1 3   // ceil(80/32)
#define NNT1 8   // 128
#define NKT2 4   // 128
#define NNT2 4   // 64

#define WPB   4        // waves per block (128 threads)
#define ACTW  288      // edge-kernel LDS activation row width (f16)
#define NACTW 128      // node-kernel LDS activation row width

__device__ __forceinline__ float silu_f(float x) { return x / (1.f + __expf(-x)); }

__device__ __forceinline__ v8f wmma16(v16h a, v16h b, v8f c) {
  return __builtin_amdgcn_wmma_f32_16x16x32_f16(false, a, false, b, (short)0, c, false, false);
}

// Load A fragment (16xK tile rows in LDS, row-major f16, stride `stride`)
// A layout (contiguous, matches packed B): lane<16: K = kb..kb+15; lane>=16: kb+16..kb+31
__device__ __forceinline__ v16h load_a(const f16* act, int stride, int row, int half, int kb) {
  v16h a;
  const f16* p0 = act + row * stride + kb + half * 16;
  const f16* p1 = p0 + 8;
  #pragma unroll
  for (int j = 0; j < 8; ++j) { a[j] = p0[j]; a[8 + j] = p1[j]; }
  return a;
}

// Load pre-packed B fragment: frag = 512 f16, lane gets contiguous 16 f16 (32B)
__device__ __forceinline__ v16h load_b(const f16* pack, int frag, int lane) {
  return *(const v16h*)(pack + (size_t)frag * 512 + lane * 16);
}

// ---------------- weight packing ----------------
// dst[((l*KT+kt)*NT+nt)*512 + lane*16 + j] = W[l][kt*32 + (lane>>4)*16 + j][nt*16 + (lane&15)]
__global__ void pack_weights(const float* __restrict__ src, f16* __restrict__ dst,
                             int K, int N, int KT, int NT) {
  long total = (long)NLAYERS * KT * NT * 512;
  for (long idx = (long)blockIdx.x * blockDim.x + threadIdx.x; idx < total;
       idx += (long)gridDim.x * blockDim.x) {
    int j = (int)(idx & 15);
    long t = idx >> 4;
    int lane = (int)(t & 31); t >>= 5;
    int nt = (int)(t % NT);  t /= NT;
    int kt = (int)(t % KT);
    int l  = (int)(t / KT);
    int n = nt * 16 + (lane & 15);
    int k = kt * 32 + ((lane >> 4) * 16) + j;
    float v = (k < K && n < N) ? src[(long)l * K * N + (long)k * N + n] : 0.f;
    dst[idx] = (f16)v;
  }
}

__global__ void init_state(const float* __restrict__ x, const float* __restrict__ pos,
                           float* __restrict__ coors, float* __restrict__ feats) {
  for (long i = (long)blockIdx.x * blockDim.x + threadIdx.x; i < (long)NN * FEATS;
       i += (long)gridDim.x * blockDim.x)
    feats[i] = x[i];
  for (long i = (long)blockIdx.x * blockDim.x + threadIdx.x; i < (long)NN * 3;
       i += (long)gridDim.x * blockDim.x)
    coors[i] = pos[i];
}

__global__ void zero_f32(float* __restrict__ p, long n) {
  for (long i = (long)blockIdx.x * blockDim.x + threadIdx.x; i < n;
       i += (long)gridDim.x * blockDim.x)
    p[i] = 0.f;
}

// ---------------- edge kernel: fused edge-MLP + coor-MLP + scatter ----------------
__global__ __launch_bounds__(128) void egnn_edge(
    const float* __restrict__ coors, const float* __restrict__ feats,
    const int* __restrict__ edges, const float* __restrict__ eattr,
    const f16* __restrict__ pWe1, const f16* __restrict__ pWe2, const f16* __restrict__ pWc1,
    const float* __restrict__ be1, const float* __restrict__ be2,
    const float* __restrict__ bc1, const float* __restrict__ Wc2, const float* __restrict__ bc2,
    float* __restrict__ m_i, float* __restrict__ mhat, int layer)
{
  __shared__ f16  s_act[WPB][16 * ACTW];
  __shared__ int  s_src[WPB][16], s_dst[WPB][16];
  __shared__ float s_rel[WPB][16][3], s_rd[WPB][16];

  const int wid  = threadIdx.x >> 5;
  const int lane = threadIdx.x & 31;
  const int ln15 = lane & 15;
  const int half = lane >> 4;
  const int ebase = (blockIdx.x * WPB + wid) * 16;
  f16* act = s_act[wid];
  const v8f zacc = {0.f,0.f,0.f,0.f,0.f,0.f,0.f,0.f};

  // 1. edge metadata (rel, |rel|^2)
  if (lane < 16) {
    int e = ebase + lane;
    int s = edges[e];
    int d = edges[NE + e];
    s_src[wid][lane] = s; s_dst[wid][lane] = d;
    float rd = 0.f;
    #pragma unroll
    for (int c = 0; c < 3; ++c) {
      float r = coors[s * 3 + c] - coors[d * 3 + c];
      s_rel[wid][lane][c] = r; rd += r * r;
    }
    s_rd[wid][lane] = rd;
  }
  __syncthreads();

  // 2. stage m_in = [feats[dst], feats[src], attr, rdist] as 16x160 (zero pad)
  {
    int sidx = s_src[wid][ln15], didx = s_dst[wid][ln15];
    const float4* rowp = (const float4*)(feats + (size_t)(half ? sidx : didx) * FEATS);
    f16* dst = act + ln15 * ACTW + half * FEATS;
    #pragma unroll
    for (int t = 0; t < 16; ++t) {
      float4 v = rowp[t];
      dst[t * 4 + 0] = (f16)v.x; dst[t * 4 + 1] = (f16)v.y;
      dst[t * 4 + 2] = (f16)v.z; dst[t * 4 + 3] = (f16)v.w;
    }
    if (half) {  // lanes 16..31: attr + rdist + zero pad (cols 128..159)
      int e = ebase + ln15;
      #pragma unroll
      for (int k = 128; k < 160; ++k) {
        float v = 0.f;
        if (k < 132)       v = eattr[e * 4 + (k - 128)];
        else if (k == 132) v = s_rd[wid][ln15];
        act[ln15 * ACTW + k] = (f16)v;
      }
    }
  }
  __syncthreads();

  // 3. GEMM1: 16x266 = m_in @ We1 (85 WMMA), nt-outer, single live accumulator
  {
    v16h afr[KT1];
    #pragma unroll
    for (int kt = 0; kt < KT1; ++kt) afr[kt] = load_a(act, ACTW, ln15, half, kt * 32);
    const f16* w1 = pWe1 + (size_t)layer * KT1 * NT1 * 512;
    const float* b1 = be1 + layer * EHID;
    __syncthreads();  // A fragments in registers; act rows may now be overwritten
    #pragma unroll
    for (int nt = 0; nt < NT1; ++nt) {
      v8f cacc = zacc;
      #pragma unroll
      for (int kt = 0; kt < KT1; ++kt)
        cacc = wmma16(afr[kt], load_b(w1, kt * NT1 + nt, lane), cacc);
      int n = nt * 16 + ln15;
      float bb = (n < EHID) ? b1[n] : 0.f;
      #pragma unroll
      for (int i = 0; i < 8; ++i)
        act[(i + 8 * half) * ACTW + n] = (f16)silu_f(cacc[i] + bb);
    }
  }
  __syncthreads();

  // 4. GEMM2: m = act @ We2 (9 WMMA)
  float msil[8];
  {
    v8f c2 = zacc;
    const f16* w2 = pWe2 + (size_t)layer * KT2 * 512;
    #pragma unroll
    for (int kt = 0; kt < KT2; ++kt) {
      v16h a = load_a(act, ACTW, ln15, half, kt * 32);
      c2 = wmma16(a, load_b(w2, kt, lane), c2);
    }
    float bb = be2[layer * MDIM + ln15];
    #pragma unroll
    for (int i = 0; i < 8; ++i) msil[i] = silu_f(c2[i] + bb);
  }
  __syncthreads();
  // stage m (16x16) at cols 0..15
  #pragma unroll
  for (int i = 0; i < 8; ++i)
    act[(i + 8 * half) * ACTW + ln15] = (f16)msil[i];
  __syncthreads();

  // 5. GEMM3: hid = m @ Wc1 (4 WMMA), SiLU, stage at cols 32..95
  {
    v16h a3 = load_a(act, ACTW, ln15, half, 0);
    const f16* w3 = pWc1 + (size_t)layer * KT3 * NT3 * 512;
    const float* b3 = bc1 + layer * 64;
    __syncthreads();
    #pragma unroll
    for (int nt = 0; nt < NT3; ++nt) {
      v8f c3 = wmma16(a3, load_b(w3, nt, lane), zacc);
      int n = nt * 16 + ln15;
      float bb = b3[n];
      #pragma unroll
      for (int i = 0; i < 8; ++i)
        act[(i + 8 * half) * ACTW + 32 + n] = (f16)silu_f(c3[i] + bb);
    }
  }
  __syncthreads();

  // 6. w = hid @ Wc2 + bc2 ; scatter w*rel into mhat (lanes 0..15)
  if (lane < 16) {
    const float* wc2 = Wc2 + layer * 64;
    float acc = bc2[layer];
    #pragma unroll 8
    for (int k = 0; k < 64; ++k)
      acc += (float)act[lane * ACTW + 32 + k] * wc2[k];
    int d = s_dst[wid][lane];
    #pragma unroll
    for (int c = 0; c < 3; ++c)
      atomicAdd(&mhat[d * 3 + c], acc * s_rel[wid][lane][c]);
  }

  // 7. scatter m into m_i (segment_sum over dst)
  #pragma unroll
  for (int i = 0; i < 8; ++i) {
    int m = i + 8 * half;
    int d = s_dst[wid][m];
    atomicAdd(&m_i[d * MDIM + ln15], msil[i]);
  }
}

// ---------------- node kernel: fused node-MLP + coors update ----------------
__global__ __launch_bounds__(128) void egnn_node(
    float* __restrict__ coors, float* __restrict__ feats,
    const float* __restrict__ m_i, const float* __restrict__ mhat,
    const f16* __restrict__ pWn1, const f16* __restrict__ pWn2,
    const float* __restrict__ bn1, const float* __restrict__ bn2, int layer)
{
  __shared__ f16 s_act[WPB][16 * NACTW];
  const int wid  = threadIdx.x >> 5;
  const int lane = threadIdx.x & 31;
  const int ln15 = lane & 15;
  const int half = lane >> 4;
  const int nbase = (blockIdx.x * WPB + wid) * 16;
  f16* act = s_act[wid];
  const v8f zacc = {0.f,0.f,0.f,0.f,0.f,0.f,0.f,0.f};

  int node = nbase + ln15;
  int nd = node < NN ? node : NN - 1;

  // 1. stage [feats | m_i] as 16x96 (cols 80..95 zero)
  {
    if (half == 0) {  // lanes 0..15: feats row (64 floats)
      const float4* rowp = (const float4*)(feats + (size_t)nd * FEATS);
      f16* dst = act + ln15 * NACTW;
      #pragma unroll
      for (int t = 0; t < 16; ++t) {
        float4 v = rowp[t];
        dst[t * 4 + 0] = (f16)v.x; dst[t * 4 + 1] = (f16)v.y;
        dst[t * 4 + 2] = (f16)v.z; dst[t * 4 + 3] = (f16)v.w;
      }
    } else {          // lanes 16..31: m_i row (16 floats) + zero pad to col 95
      const float4* rowp = (const float4*)(m_i + (size_t)nd * MDIM);
      f16* dst = act + ln15 * NACTW + FEATS;
      #pragma unroll
      for (int t = 0; t < 4; ++t) {
        float4 v = rowp[t];
        dst[t * 4 + 0] = (f16)v.x; dst[t * 4 + 1] = (f16)v.y;
        dst[t * 4 + 2] = (f16)v.z; dst[t * 4 + 3] = (f16)v.w;
      }
      #pragma unroll
      for (int k = 16; k < 32; ++k) dst[k] = (f16)0.f;
    }
  }
  __syncthreads();

  // 2. GEMM1: u = [feats|m_i] @ Wn1 (24 WMMA), nt-outer
  {
    v16h afr[NKT1];
    #pragma unroll
    for (int kt = 0; kt < NKT1; ++kt) afr[kt] = load_a(act, NACTW, ln15, half, kt * 32);
    const f16* w1 = pWn1 + (size_t)layer * NKT1 * NNT1 * 512;
    const float* b1 = bn1 + layer * 128;
    __syncthreads();
    #pragma unroll
    for (int nt = 0; nt < NNT1; ++nt) {
      v8f cacc = zacc;
      #pragma unroll
      for (int kt = 0; kt < NKT1; ++kt)
        cacc = wmma16(afr[kt], load_b(w1, kt * NNT1 + nt, lane), cacc);
      int n = nt * 16 + ln15;
      float bb = b1[n];
      #pragma unroll
      for (int i = 0; i < 8; ++i)
        act[(i + 8 * half) * NACTW + n] = (f16)silu_f(cacc[i] + bb);
    }
  }
  __syncthreads();

  // 3. GEMM2: du = u @ Wn2 (16 WMMA), nt-outer with preloaded A
  {
    v16h afr[NKT2];
    #pragma unroll
    for (int kt = 0; kt < NKT2; ++kt) afr[kt] = load_a(act, NACTW, ln15, half, kt * 32);
    const f16* w2 = pWn2 + (size_t)layer * NKT2 * NNT2 * 512;
    const float* b2 = bn2 + layer * 64;
    #pragma unroll
    for (int nt = 0; nt < NNT2; ++nt) {
      v8f cacc = zacc;
      #pragma unroll
      for (int kt = 0; kt < NKT2; ++kt)
        cacc = wmma16(afr[kt], load_b(w2, kt * NNT2 + nt, lane), cacc);
      int n = nt * 16 + ln15;
      float bb = b2[n];
      #pragma unroll
      for (int i = 0; i < 8; ++i) {
        int m = i + 8 * half;
        int nn = nbase + m;
        if (nn < NN) feats[nn * FEATS + n] += cacc[i] + bb;  // residual
      }
    }
  }

  // 4. coors += mhat
  if (lane < 16) {
    int nn = nbase + lane;
    if (nn < NN) {
      #pragma unroll
      for (int cc = 0; cc < 3; ++cc)
        coors[nn * 3 + cc] += mhat[nn * 3 + cc];
    }
  }
}

__global__ void write_out(const float* __restrict__ coors, const float* __restrict__ feats,
                          float* __restrict__ out) {
  for (long i = (long)blockIdx.x * blockDim.x + threadIdx.x; i < (long)NN * 67;
       i += (long)gridDim.x * blockDim.x) {
    int n = (int)(i / 67), cix = (int)(i % 67);
    out[i] = (cix < 3) ? coors[n * 3 + cix] : feats[n * FEATS + (cix - 3)];
  }
}

extern "C" void kernel_launch(void* const* d_in, const int* in_sizes, int n_in,
                              void* d_out, int out_size, void* d_ws, size_t ws_size,
                              hipStream_t stream) {
  const float* x     = (const float*)d_in[0];
  const float* pos   = (const float*)d_in[1];
  const int*   edges = (const int*)  d_in[2];
  const float* eatt  = (const float*)d_in[3];
  const float* We1   = (const float*)d_in[4];
  const float* be1   = (const float*)d_in[5];
  const float* We2   = (const float*)d_in[6];
  const float* be2   = (const float*)d_in[7];
  const float* Wc1   = (const float*)d_in[8];
  const float* bc1   = (const float*)d_in[9];
  const float* Wc2   = (const float*)d_in[10];
  const float* bc2   = (const float*)d_in[11];
  const float* Wn1   = (const float*)d_in[12];
  const float* bn1   = (const float*)d_in[13];
  const float* Wn2   = (const float*)d_in[14];
  const float* bn2   = (const float*)d_in[15];

  // workspace carve-up (256B aligned)
  size_t off = 0;
  auto carve = [&](size_t bytes) { size_t o = off; off = (off + bytes + 255) & ~(size_t)255; return o; };
  char* ws = (char*)d_ws;
  float* coors = (float*)(ws + carve((size_t)NN * 3 * 4));
  float* feats = (float*)(ws + carve((size_t)NN * FEATS * 4));
  float* m_i   = (float*)(ws + carve((size_t)NN * MDIM * 4));
  float* mhat  = (float*)(ws + carve((size_t)NN * 3 * 4));
  f16* pWe1 = (f16*)(ws + carve((size_t)NLAYERS * KT1 * NT1 * 512 * 2));
  f16* pWe2 = (f16*)(ws + carve((size_t)NLAYERS * KT2 * 1 * 512 * 2));
  f16* pWc1 = (f16*)(ws + carve((size_t)NLAYERS * KT3 * NT3 * 512 * 2));
  f16* pWn1 = (f16*)(ws + carve((size_t)NLAYERS * NKT1 * NNT1 * 512 * 2));
  f16* pWn2 = (f16*)(ws + carve((size_t)NLAYERS * NKT2 * NNT2 * 512 * 2));
  (void)ws_size; (void)in_sizes; (void)n_in; (void)out_size;

  // pack weights -> WMMA B-fragment layout (f16, zero padded)
  pack_weights<<<512, 256, 0, stream>>>(We1, pWe1, EIN,  EHID, KT1,  NT1);
  pack_weights<<<128, 256, 0, stream>>>(We2, pWe2, EHID, MDIM, KT2,  1);
  pack_weights<<<64,  256, 0, stream>>>(Wc1, pWc1, MDIM, 64,   KT3,  NT3);
  pack_weights<<<128, 256, 0, stream>>>(Wn1, pWn1, FEATS + MDIM, 128, NKT1, NNT1);
  pack_weights<<<128, 256, 0, stream>>>(Wn2, pWn2, 128,  FEATS, NKT2, NNT2);

  init_state<<<1024, 256, 0, stream>>>(x, pos, coors, feats);

  const int edge_blocks = NE / 16 / WPB;                  // 6250
  const int node_blocks = (NN / 16 + WPB - 1) / WPB;      // 782 (3128 tiles >= 3125)
  for (int l = 0; l < NLAYERS; ++l) {
    zero_f32<<<512, 256, 0, stream>>>(m_i,  (long)NN * MDIM);
    zero_f32<<<256, 256, 0, stream>>>(mhat, (long)NN * 3);
    egnn_edge<<<edge_blocks, 32 * WPB, 0, stream>>>(
        coors, feats, edges, eatt, pWe1, pWe2, pWc1,
        be1, be2, bc1, Wc2, bc2, m_i, mhat, l);
    egnn_node<<<node_blocks, 32 * WPB, 0, stream>>>(
        coors, feats, m_i, mhat, pWn1, pWn2, bn1, bn2, l);
  }
  write_out<<<1024, 256, 0, stream>>>(coors, feats, (float*)d_out);
}
